// Graph_HiC_Likelihood_3092376453283
// MI455X (gfx1250) — matrix operand order repaired and checked by reading, not compiled, not run on hardware
//
#include <hip/hip_runtime.h>
#include <hip/hip_bf16.h>

// ---------------- problem constants ----------------
enum : int {
  NNODES = 50000,
  NEDGES = 800000,
  NTEST  = 200000,
  INDIM  = 74,
  INPAD  = 96,                  // INDIM padded to multiple of 32
  HID    = 128,
  NHEADS = 4,
  NEAR   = 5,
  OUTDIM = 4,
  LIKD   = 128,
  GATW   = NHEADS * HID,        // 512
  GCNW   = NEAR * HID,          // 640
  NELOOP = NEDGES + NNODES      // 850000, edges incl. self loops
};

// ---------------- WMMA types ----------------
typedef __attribute__((ext_vector_type(16))) __bf16 bf16x16;
typedef __attribute__((ext_vector_type(8)))  __bf16 bf16x8;
typedef __attribute__((ext_vector_type(8)))  float  f32x8;

__device__ __forceinline__ __bf16 fb(float v) { return (__bf16)v; }

// monotone float<->uint encoding for atomic segment-max
__device__ __forceinline__ unsigned fenc(float f) {
  unsigned u = __float_as_uint(f);
  return u ^ ((u >> 31) ? 0xFFFFFFFFu : 0x80000000u);
}
__device__ __forceinline__ float fdec(unsigned u) {
  unsigned b = (u & 0x80000000u) ? (u ^ 0x80000000u) : ~u;
  return __uint_as_float(b);
}

// =====================================================================
// Pack fp32 weight B[K x Nout] into bf16 WMMA-fragment layout:
//   P[kblk][half][n][i] , element = B[kblk*32 + half*16 + i][n]  (0 if k>=Ksrc)
// Lane (n, half) then reads 16 contiguous bf16 (= 2 x b128) per K-step.
// =====================================================================
__global__ void pack_b_kernel(const float* __restrict__ B, __bf16* __restrict__ P,
                              int Ksrc, int Kpad, int Nout) {
  long j = blockIdx.x * (long)blockDim.x + threadIdx.x;
  long total = (long)Kpad * Nout;
  if (j >= total) return;
  int  i  = (int)(j & 15);
  long r  = j >> 4;
  int  n  = (int)(r % Nout);
  long r2 = r / Nout;
  int  h  = (int)(r2 & 1);
  int  kb = (int)(r2 >> 1);
  int  k  = kb * 32 + h * 16 + i;
  float v = (k < Ksrc) ? B[(size_t)k * Nout + n] : 0.0f;
  P[j] = fb(v);
}

// zero-pad node features x (N x 74) -> xpad (N x 96)
__global__ void pad_x_kernel(const float* __restrict__ x, float* __restrict__ xp) {
  long j = blockIdx.x * (long)blockDim.x + threadIdx.x;
  if (j >= (long)NNODES * INPAD) return;
  int n = (int)(j / INPAD), k = (int)(j % INPAD);
  xp[j] = (k < INDIM) ? x[(size_t)n * INDIM + k] : 0.0f;
}

// =====================================================================
// WMMA GEMM, branch-free inner loop. K must be a multiple of 32.
// GATHER: A row m = concat(H[idxU[m]], H[idxV[m]]), halfK%32==0 so each
// K-block comes entirely from one half (select hoisted out of elem loop).
// =====================================================================
template <bool GATHER>
__global__ void __launch_bounds__(256)
wmma_gemm_pk(const float* __restrict__ A, int lda,
             const int* __restrict__ idxU, const int* __restrict__ idxV, int halfK,
             const __bf16* __restrict__ Bp,
             const float* __restrict__ bias,
             float* __restrict__ C, int ldc, int coff,
             int M, int K, int Nout, int relu)
{
  const int lane   = threadIdx.x & 31;
  const int waveId = blockIdx.x * 8 + (threadIdx.x >> 5);
  const int ntile  = Nout >> 4;
  const int mt = waveId / ntile;
  const int nt = waveId - mt * ntile;
  if (mt * 16 >= M) return;                  // wave-uniform exit

  const int laneLo = lane & 15;
  const int laneHi = lane >> 4;              // 0 or 1
  int m = mt * 16 + laneLo;
  if (m >= M) m = M - 1;                     // clamp; OOB rows discarded at store
  const int n = nt * 16 + laneLo;

  const float* __restrict__ rowU;
  const float* __restrict__ rowV = nullptr;
  if (GATHER) {
    rowU = A + (size_t)idxU[m] * halfK;
    rowV = A + (size_t)idxV[m] * halfK;
  } else {
    rowU = A + (size_t)m * lda;
  }

  const __bf16* __restrict__ bp = Bp + ((size_t)laneHi * Nout + n) * 16;
  const size_t bstep = (size_t)Nout * 32;

  f32x8 acc = {};
  for (int k0 = 0; k0 < K; k0 += 32) {
    const float* src;
    int kbase;
    if (GATHER) {
      bool lo = (k0 < halfK);
      src   = lo ? rowU : rowV;
      kbase = lo ? k0 : (k0 - halfK);
    } else {
      src   = rowU;
      kbase = k0;
    }
    const float* p = src + kbase + laneHi * 16;
    __builtin_prefetch((const void*)(p + 32), 0, 3);   // next K-step (global_prefetch)
    float4 a0 = *(const float4*)(p);
    float4 a1 = *(const float4*)(p + 4);
    float4 a2 = *(const float4*)(p + 8);
    float4 a3 = *(const float4*)(p + 12);
    bf16x16 af;
    af[0]  = fb(a0.x); af[1]  = fb(a0.y); af[2]  = fb(a0.z); af[3]  = fb(a0.w);
    af[4]  = fb(a1.x); af[5]  = fb(a1.y); af[6]  = fb(a1.z); af[7]  = fb(a1.w);
    af[8]  = fb(a2.x); af[9]  = fb(a2.y); af[10] = fb(a2.z); af[11] = fb(a2.w);
    af[12] = fb(a3.x); af[13] = fb(a3.y); af[14] = fb(a3.z); af[15] = fb(a3.w);

    bf16x8 blo = *(const bf16x8*)(bp);
    bf16x8 bhi = *(const bf16x8*)(bp + 8);
    bf16x16 bfv = __builtin_shufflevector(blo, bhi,
        0, 1, 2, 3, 4, 5, 6, 7, 8, 9, 10, 11, 12, 13, 14, 15);
    bp += bstep;

    acc = __builtin_amdgcn_wmma_f32_16x16x32_bf16(
        false, af, false, bfv, (short)0, acc, false, false);
  }

  const float bv = bias ? bias[n] : 0.0f;
#pragma unroll
  for (int r = 0; r < 8; ++r) {
    int mm = mt * 16 + laneHi * 8 + r;       // C layout: vgpr r -> row r (+8 upper half)
    if (mm < M) {
      float v = acc[r] + bv;
      if (relu) v = fmaxf(v, 0.0f);
      C[(size_t)mm * ldc + coff + n] = v;
    }
  }
}

// =====================================================================
// fills
// =====================================================================
__global__ void fill_f32_kernel(float* __restrict__ p, float v, long n) {
  long j = blockIdx.x * (long)blockDim.x + threadIdx.x;
  if (j < n) p[j] = v;
}
__global__ void fill_u32_kernel(unsigned* __restrict__ p, unsigned v, long n) {
  long j = blockIdx.x * (long)blockDim.x + threadIdx.x;
  if (j < n) p[j] = v;
}

// =====================================================================
// graph prep
// =====================================================================
__global__ void build_edges_kernel(const int* __restrict__ ei,
                                   int* __restrict__ row, int* __restrict__ col) {
  int e = blockIdx.x * blockDim.x + threadIdx.x;
  if (e >= NELOOP) return;
  if (e < NEDGES) { row[e] = ei[e]; col[e] = ei[NEDGES + e]; }
  else            { row[e] = e - NEDGES; col[e] = e - NEDGES; }
}

__global__ void deg_kernel(const int* __restrict__ col,
                           const float* __restrict__ attr,
                           float* __restrict__ deg) {
  int e = blockIdx.x * blockDim.x + threadIdx.x;
  if (e >= NELOOP) return;
  int c = col[e];
#pragma unroll
  for (int i = 0; i < NEAR; ++i) {
    float w = (e < NEDGES) ? attr[(size_t)e * NEAR + i] : 1.0f;
    atomicAdd(&deg[(size_t)i * NNODES + c], w);
  }
}

__global__ void dis_kernel(float* __restrict__ deg) {
  long j = blockIdx.x * (long)blockDim.x + threadIdx.x;
  if (j >= (long)NEAR * NNODES) return;
  float d = deg[j];
  deg[j] = (d > 0.0f) ? rsqrtf(fmaxf(d, 1e-30f)) : 0.0f;
}

__global__ void norm_kernel(const int* __restrict__ row, const int* __restrict__ col,
                            const float* __restrict__ attr,
                            const float* __restrict__ dis,
                            float* __restrict__ norm) {
  int e = blockIdx.x * blockDim.x + threadIdx.x;
  if (e >= NELOOP) return;
  int r = row[e], c = col[e];
#pragma unroll
  for (int i = 0; i < NEAR; ++i) {
    float w = (e < NEDGES) ? attr[(size_t)e * NEAR + i] : 1.0f;
    norm[(size_t)i * NELOOP + e] =
        dis[(size_t)i * NNODES + r] * w * dis[(size_t)i * NNODES + c];
  }
}

// =====================================================================
// GAT attention
// =====================================================================
__global__ void att_score_kernel(const float* __restrict__ hlin,
                                 const float* __restrict__ asrc,
                                 const float* __restrict__ adst,
                                 float* __restrict__ a_s, float* __restrict__ a_d) {
  int j = blockIdx.x * blockDim.x + threadIdx.x;
  if (j >= NNODES * NHEADS) return;
  int n = j / NHEADS, hd = j % NHEADS;
  const float* hp = hlin + (size_t)n * GATW + hd * HID;
  float s = 0.0f, d = 0.0f;
  for (int c = 0; c < HID; ++c) {
    float v = hp[c];
    s += v * asrc[hd * HID + c];
    d += v * adst[hd * HID + c];
  }
  a_s[j] = s; a_d[j] = d;
}

__global__ void att_max_kernel(const int* __restrict__ row, const int* __restrict__ col,
                               const float* __restrict__ a_s, const float* __restrict__ a_d,
                               unsigned* __restrict__ menc) {
  int e = blockIdx.x * blockDim.x + threadIdx.x;
  if (e >= NELOOP) return;
  int r = row[e], c = col[e];
#pragma unroll
  for (int hd = 0; hd < NHEADS; ++hd) {
    float ev = a_s[r * NHEADS + hd] + a_d[c * NHEADS + hd];
    ev = (ev >= 0.0f) ? ev : 0.2f * ev;               // leaky_relu
    atomicMax(&menc[c * NHEADS + hd], fenc(ev));
  }
}

__global__ void att_exp_kernel(const int* __restrict__ row, const int* __restrict__ col,
                               const float* __restrict__ a_s, const float* __restrict__ a_d,
                               const unsigned* __restrict__ menc,
                               float* __restrict__ alpha, float* __restrict__ denom) {
  int e = blockIdx.x * blockDim.x + threadIdx.x;
  if (e >= NELOOP) return;
  int r = row[e], c = col[e];
#pragma unroll
  for (int hd = 0; hd < NHEADS; ++hd) {
    float ev = a_s[r * NHEADS + hd] + a_d[c * NHEADS + hd];
    ev = (ev >= 0.0f) ? ev : 0.2f * ev;
    float m = fdec(menc[c * NHEADS + hd]);
    float p = expf(ev - m);
    alpha[(size_t)e * NHEADS + hd] = p;
    atomicAdd(&denom[c * NHEADS + hd], p);
  }
}

__global__ void gat_scatter_kernel(const int* __restrict__ row, const int* __restrict__ col,
                                   const float* __restrict__ alpha,
                                   const float* __restrict__ denom,
                                   const float* __restrict__ hlin,
                                   float* __restrict__ hout) {
  long idx = blockIdx.x * (long)blockDim.x + threadIdx.x;
  if (idx >= (long)NELOOP * HID) return;
  int e = (int)(idx >> 7);
  int d = (int)(idx & 127);
  int r = row[e], c = col[e];
#pragma unroll
  for (int hd = 0; hd < NHEADS; ++hd) {
    float coef = alpha[(size_t)e * NHEADS + hd] /
                 (denom[c * NHEADS + hd] + 1e-16f);
    atomicAdd(&hout[(size_t)c * GATW + hd * HID + d],
              coef * hlin[(size_t)r * GATW + hd * HID + d]);
  }
}

__global__ void bias_relu_kernel(float* __restrict__ h, const float* __restrict__ b,
                                 int width, long total) {
  long j = blockIdx.x * (long)blockDim.x + threadIdx.x;
  if (j >= total) return;
  int c = (int)(j % width);
  h[j] = fmaxf(h[j] + b[c], 0.0f);
}

// =====================================================================
// GCN norm-weighted scatter: out[col, coff+d] += norm[e] * t[row, d]
// =====================================================================
__global__ void gcn_scatter_kernel(const int* __restrict__ row, const int* __restrict__ col,
                                   const float* __restrict__ normi,
                                   const float* __restrict__ t,
                                   float* __restrict__ out, int coff) {
  long idx = blockIdx.x * (long)blockDim.x + threadIdx.x;
  if (idx >= (long)NELOOP * HID) return;
  int e = (int)(idx >> 7);
  int d = (int)(idx & 127);
  float nv = normi[e];
  atomicAdd(&out[(size_t)col[e] * GCNW + coff + d],
            nv * t[(size_t)row[e] * HID + d]);
}

// =====================================================================
// link MLP final layer (128 -> 4) + symmetric combine with [0,2,1,3] perm
// =====================================================================
__global__ void mlp_out_kernel(const float* __restrict__ h4,
                               const float* __restrict__ W,
                               const float* __restrict__ b,
                               float* __restrict__ out, int pass) {
  int r = blockIdx.x * blockDim.x + threadIdx.x;
  if (r >= NTEST) return;
  const int perm[4] = {0, 2, 1, 3};
#pragma unroll
  for (int o = 0; o < OUTDIM; ++o) {
    float s = b[o];
    for (int c = 0; c < LIKD; ++c)
      s += h4[(size_t)r * LIKD + c] * W[c * OUTDIM + o];
    if (pass == 0) out[(size_t)r * OUTDIM + o]        = 0.5f * s;
    else           out[(size_t)r * OUTDIM + perm[o]] += 0.5f * s;
  }
}

// =====================================================================
// host orchestration
// =====================================================================
static inline unsigned gridFor(long n, int t) { return (unsigned)((n + t - 1) / t); }

static void launch_gemm(bool gather, const float* A, int lda,
                        const int* iu, const int* iv, int halfK,
                        const __bf16* Bp, const float* bias, float* C,
                        int ldc, int coff, int M, int K, int Nout, int relu,
                        hipStream_t s) {
  long waves = (long)((M + 15) / 16) * (Nout / 16);
  unsigned blocks = (unsigned)((waves + 7) / 8);
  if (gather)
    wmma_gemm_pk<true><<<blocks, 256, 0, s>>>(A, lda, iu, iv, halfK, Bp, bias,
                                              C, ldc, coff, M, K, Nout, relu);
  else
    wmma_gemm_pk<false><<<blocks, 256, 0, s>>>(A, lda, iu, iv, halfK, Bp, bias,
                                               C, ldc, coff, M, K, Nout, relu);
}

extern "C" void kernel_launch(void* const* d_in, const int* in_sizes, int n_in,
                              void* d_out, int out_size, void* d_ws, size_t ws_size,
                              hipStream_t stream) {
  const float* x        = (const float*)d_in[0];
  const int*   ei       = (const int*)d_in[1];
  const float* attr     = (const float*)d_in[2];
  const int*   eit      = (const int*)d_in[3];
  const float* gat_W    = (const float*)d_in[4];
  const float* att_src  = (const float*)d_in[5];
  const float* att_dst  = (const float*)d_in[6];
  const float* gat_b    = (const float*)d_in[7];
  const float* gcn0_W   = (const float*)d_in[8];
  const float* gcn0_b   = (const float*)d_in[9];
  const float* gcn_W    = (const float*)d_in[10];
  const float* gcn_b    = (const float*)d_in[11];
  const float* lk_W[5]  = {(const float*)d_in[12], (const float*)d_in[14],
                           (const float*)d_in[16], (const float*)d_in[18],
                           (const float*)d_in[20]};
  const float* lk_b[5]  = {(const float*)d_in[13], (const float*)d_in[15],
                           (const float*)d_in[17], (const float*)d_in[19],
                           (const float*)d_in[21]};
  float* out = (float*)d_out;

  // ---- workspace layout ----
  char* ws = (char*)d_ws;
  size_t off = 0;
  auto wsAlloc = [&](size_t bytes) -> char* {
    char* p = ws + off;
    off = (off + bytes + 255) & ~(size_t)255;
    return p;
  };
  int*      rowv   = (int*)wsAlloc((size_t)NELOOP * 4);
  int*      colv   = (int*)wsAlloc((size_t)NELOOP * 4);
  float*    deg    = (float*)wsAlloc((size_t)NEAR * NNODES * 4);   // becomes dis
  float*    norm   = (float*)wsAlloc((size_t)NEAR * NELOOP * 4);
  float*    xpad   = (float*)wsAlloc((size_t)NNODES * INPAD * 4);
  float*    hlin   = (float*)wsAlloc((size_t)NNODES * GATW * 4);
  float*    hattn  = (float*)wsAlloc((size_t)NNODES * GATW * 4);
  float*    a_s    = (float*)wsAlloc((size_t)NNODES * NHEADS * 4);
  float*    a_d    = (float*)wsAlloc((size_t)NNODES * NHEADS * 4);
  unsigned* menc   = (unsigned*)wsAlloc((size_t)NNODES * NHEADS * 4);
  float*    denom  = (float*)wsAlloc((size_t)NNODES * NHEADS * 4);
  float*    alpha  = (float*)wsAlloc((size_t)NELOOP * NHEADS * 4);
  float*    hA     = (float*)wsAlloc((size_t)NNODES * GCNW * 4);
  float*    hB     = (float*)wsAlloc((size_t)NNODES * GCNW * 4);
  float*    tbuf   = (float*)wsAlloc((size_t)NNODES * HID * 4);
  float*    m1     = (float*)wsAlloc((size_t)NTEST * LIKD * 4);
  float*    m2     = (float*)wsAlloc((size_t)NTEST * LIKD * 4);
  __bf16*   pk_gat  = (__bf16*)wsAlloc((size_t)INPAD * GATW * 2);
  __bf16*   pk_gcn0 = (__bf16*)wsAlloc((size_t)NEAR * GATW * HID * 2);
  __bf16*   pk_gcn  = (__bf16*)wsAlloc((size_t)20 * GCNW * HID * 2);
  __bf16*   pk_lk0  = (__bf16*)wsAlloc((size_t)2 * GCNW * LIKD * 2);
  __bf16*   pk_lk1  = (__bf16*)wsAlloc((size_t)LIKD * LIKD * 2);
  __bf16*   pk_lk2  = (__bf16*)wsAlloc((size_t)LIKD * LIKD * 2);
  __bf16*   pk_lk3  = (__bf16*)wsAlloc((size_t)LIKD * LIKD * 2);
  (void)ws_size; (void)in_sizes; (void)n_in; (void)out_size;

  const int T = 256;

  // ---- 0. pack all weights to bf16 fragment layout; pad x ----
  auto packB = [&](const float* B, __bf16* P, int Ksrc, int Kpad, int Nout) {
    long total = (long)Kpad * Nout;
    pack_b_kernel<<<gridFor(total, T), T, 0, stream>>>(B, P, Ksrc, Kpad, Nout);
  };
  packB(gat_W, pk_gat, INDIM, INPAD, GATW);
  for (int i = 0; i < NEAR; ++i)
    packB(gcn0_W + (size_t)i * GATW * HID, pk_gcn0 + (size_t)i * GATW * HID,
          GATW, GATW, HID);
  for (int j = 0; j < 20; ++j)
    packB(gcn_W + (size_t)j * GCNW * HID, pk_gcn + (size_t)j * GCNW * HID,
          GCNW, GCNW, HID);
  packB(lk_W[0], pk_lk0, 2 * GCNW, 2 * GCNW, LIKD);
  packB(lk_W[1], pk_lk1, LIKD, LIKD, LIKD);
  packB(lk_W[2], pk_lk2, LIKD, LIKD, LIKD);
  packB(lk_W[3], pk_lk3, LIKD, LIKD, LIKD);
  pad_x_kernel<<<gridFor((long)NNODES * INPAD, T), T, 0, stream>>>(x, xpad);

  // ---- 1. graph prep ----
  build_edges_kernel<<<gridFor(NELOOP, T), T, 0, stream>>>(ei, rowv, colv);
  fill_f32_kernel<<<gridFor((long)NEAR * NNODES, T), T, 0, stream>>>(deg, 0.0f, (long)NEAR * NNODES);
  deg_kernel<<<gridFor(NELOOP, T), T, 0, stream>>>(colv, attr, deg);
  dis_kernel<<<gridFor((long)NEAR * NNODES, T), T, 0, stream>>>(deg);
  norm_kernel<<<gridFor(NELOOP, T), T, 0, stream>>>(rowv, colv, attr, deg, norm);

  // ---- 2. GAT ----
  launch_gemm(false, xpad, INPAD, nullptr, nullptr, 0, pk_gat, nullptr,
              hlin, GATW, 0, NNODES, INPAD, GATW, 0, stream);
  att_score_kernel<<<gridFor(NNODES * NHEADS, T), T, 0, stream>>>(hlin, att_src, att_dst, a_s, a_d);
  fill_u32_kernel<<<gridFor((long)NNODES * NHEADS, T), T, 0, stream>>>(menc, 0x007FFFFFu, (long)NNODES * NHEADS); // enc(-inf)
  att_max_kernel<<<gridFor(NELOOP, T), T, 0, stream>>>(rowv, colv, a_s, a_d, menc);
  fill_f32_kernel<<<gridFor((long)NNODES * NHEADS, T), T, 0, stream>>>(denom, 0.0f, (long)NNODES * NHEADS);
  att_exp_kernel<<<gridFor(NELOOP, T), T, 0, stream>>>(rowv, colv, a_s, a_d, menc, alpha, denom);
  fill_f32_kernel<<<gridFor((long)NNODES * GATW, T), T, 0, stream>>>(hattn, 0.0f, (long)NNODES * GATW);
  gat_scatter_kernel<<<gridFor((long)NELOOP * HID, T), T, 0, stream>>>(rowv, colv, alpha, denom, hlin, hattn);
  bias_relu_kernel<<<gridFor((long)NNODES * GATW, T), T, 0, stream>>>(hattn, gat_b, GATW, (long)NNODES * GATW);

  // ---- 3. gcn0: 512 -> 5x128 ----
  fill_f32_kernel<<<gridFor((long)NNODES * GCNW, T), T, 0, stream>>>(hA, 0.0f, (long)NNODES * GCNW);
  for (int i = 0; i < NEAR; ++i) {
    launch_gemm(false, hattn, GATW, nullptr, nullptr, 0,
                pk_gcn0 + (size_t)i * GATW * HID, nullptr,
                tbuf, HID, 0, NNODES, GATW, HID, 0, stream);
    gcn_scatter_kernel<<<gridFor((long)NELOOP * HID, T), T, 0, stream>>>(
        rowv, colv, norm + (size_t)i * NELOOP, tbuf, hA, i * HID);
  }
  bias_relu_kernel<<<gridFor((long)NNODES * GCNW, T), T, 0, stream>>>(hA, gcn0_b, GCNW, (long)NNODES * GCNW);

  // ---- 4. gcn layers 0..3: 640 -> 5x128, ping-pong hA/hB ----
  float* hin = hA; float* hout = hB;
  for (int l = 0; l < 4; ++l) {
    fill_f32_kernel<<<gridFor((long)NNODES * GCNW, T), T, 0, stream>>>(hout, 0.0f, (long)NNODES * GCNW);
    for (int i = 0; i < NEAR; ++i) {
      launch_gemm(false, hin, GCNW, nullptr, nullptr, 0,
                  pk_gcn + (size_t)(l * NEAR + i) * GCNW * HID, nullptr,
                  tbuf, HID, 0, NNODES, GCNW, HID, 0, stream);
      gcn_scatter_kernel<<<gridFor((long)NELOOP * HID, T), T, 0, stream>>>(
          rowv, colv, norm + (size_t)i * NELOOP, tbuf, hout, i * HID);
    }
    bias_relu_kernel<<<gridFor((long)NNODES * GCNW, T), T, 0, stream>>>(
        hout, gcn_b + (size_t)l * GCNW, GCNW, (long)NNODES * GCNW);
    float* tmp = hin; hin = hout; hout = tmp;
  }
  const float* hfin = hin;   // final node embeddings (N, 640)

  // ---- 5. link MLP, two symmetric passes ----
  for (int pass = 0; pass < 2; ++pass) {
    const int* U = (pass == 0) ? eit : (eit + NTEST);
    const int* V = (pass == 0) ? (eit + NTEST) : eit;
    // layer 0: gather-concat (1280 -> 128), relu
    launch_gemm(true, hfin, 0, U, V, GCNW, pk_lk0, lk_b[0],
                m1, LIKD, 0, NTEST, 2 * GCNW, LIKD, 1, stream);
    // layer 1: relu
    launch_gemm(false, m1, LIKD, nullptr, nullptr, 0, pk_lk1, lk_b[1],
                m2, LIKD, 0, NTEST, LIKD, LIKD, 1, stream);
    // layer 2: no relu
    launch_gemm(false, m2, LIKD, nullptr, nullptr, 0, pk_lk2, lk_b[2],
                m1, LIKD, 0, NTEST, LIKD, LIKD, 0, stream);
    // layer 3: relu
    launch_gemm(false, m1, LIKD, nullptr, nullptr, 0, pk_lk3, lk_b[3],
                m2, LIKD, 0, NTEST, LIKD, LIKD, 1, stream);
    // layer 4 + combine
    mlp_out_kernel<<<gridFor(NTEST, T), T, 0, stream>>>(m2, lk_W[4], lk_b[4], out, pass);
  }
}
